// MultiTaskGNN_43009802502560
// MI455X (gfx1250) — matrix operand order repaired and checked by reading, not compiled, not run on hardware
//
#include <hip/hip_runtime.h>
#include <math.h>

// MultiTaskGNN (2-layer GATv2) for MI455X / gfx1250, wave32.
// GEMMs: v_wmma_f32_16x16x32_f16, 16x32 tile/wave, fully unrolled K (templated),
// weights pre-repacked to column-major f16 so B-fragments are contiguous b128 loads.
// Edge stage (memory-bound): wave-per-edge, lane==channel, native f32 atomics.

typedef __attribute__((ext_vector_type(16))) _Float16 v16h;
typedef __attribute__((ext_vector_type(8)))  float    v8f;

#define NEG_SLOPE 0.2f

// ---------- small helpers ----------

__device__ __forceinline__ void atomicMaxF(float* addr, float v) {
  // sign-split trick: lowers to global_atomic_max_i32 / global_atomic_min_u32
  if (v >= 0.0f) atomicMax((int*)addr, __float_as_int(v));
  else           atomicMin((unsigned int*)addr, __float_as_uint(v));
}

__global__ void fill_kernel(float* __restrict__ p, float v, int n) {
  int i = blockIdx.x * blockDim.x + threadIdx.x;
  if (i < n) p[i] = v;
}

// ---------- preprocessing: degree + mean edge_attr per dst (self-loop fill) ----------

__global__ void deg_easum_kernel(const int* __restrict__ dst, const float* __restrict__ ea,
                                 float* __restrict__ deg, float* __restrict__ easum, int E) {
  int e = blockIdx.x * blockDim.x + threadIdx.x;
  if (e >= E) return;
  int d = dst[e];
  atomicAdd(&deg[d], 1.0f);
  atomicAdd(&easum[d], ea[e]);
}

__global__ void eamean_kernel(float* __restrict__ easum, const float* __restrict__ deg, int n) {
  int i = blockIdx.x * blockDim.x + threadIdx.x;
  if (i < n) easum[i] /= fmaxf(deg[i], 1.0f);
}

// ---------- weight repack: W[K x Nc] f32 row-major -> Wt[Nc x K] f16 (col-major of W) ----------

__global__ void repack_w_kernel(const float* __restrict__ W, _Float16* __restrict__ Wt,
                                int K, int Nc) {
  int i = blockIdx.x * blockDim.x + threadIdx.x;
  if (i >= K * Nc) return;
  int row = i / Nc, col = i - row * Nc;
  Wt[col * K + row] = (_Float16)W[i];
}

// ---------- WMMA GEMM: C[M x NC] = A[M x K](f32->f16) * W(f16,repacked) + bias ----------
// One 16x32 output tile per wave (two accumulators share the A fragment).
// Requires M % 16 == 0 (holds: M = 50000). A/B fragments use contiguous k order (k = 16*half + s) in both operands.

template <int K, int NC>
__global__ void gemm_bias_wmma(const float* __restrict__ A, const _Float16* __restrict__ Wt,
                               const float* __restrict__ bias, float* __restrict__ C, int M) {
  const int wid  = threadIdx.x >> 5;
  const int lane = threadIdx.x & 31;
  constexpr int ntn = NC / 32;                         // 32-wide column tiles
  const int tiles_total = (M >> 4) * ntn;
  int t = blockIdx.x * (blockDim.x >> 5) + wid;
  if (t >= tiles_total) return;
  const int tm    = t / ntn;
  const int tn    = t - tm * ntn;
  const int half  = lane >> 4;                         // 0: lanes 0-15, 1: lanes 16-31
  const int l16   = lane & 15;
  const int rowA  = (tm << 4) + l16;                   // A-frag: lane holds one M row
  const int colB0 = (tn << 5) + l16;                   // B-frag: lane holds one N column
  const int colB1 = colB0 + 16;

  const float*    ap  = A  + (size_t)rowA  * K + half * 16;
  const _Float16* bp0 = Wt + (size_t)colB0 * K + half * 16;
  const _Float16* bp1 = Wt + (size_t)colB1 * K + half * 16;

  v8f acc0 = {}, acc1 = {};
#pragma unroll
  for (int kk = 0; kk < K; kk += 32) {
    // A 16x32 f16: halves 0-15 = K kk+half*16 .. +15
    const float4 a0 = *(const float4*)(ap + kk);
    const float4 a1 = *(const float4*)(ap + kk + 4);
    const float4 a2 = *(const float4*)(ap + kk + 8);
    const float4 a3 = *(const float4*)(ap + kk + 12);
    v16h a;
    a[0]  = (_Float16)a0.x; a[1]  = (_Float16)a0.y; a[2]  = (_Float16)a0.z; a[3]  = (_Float16)a0.w;
    a[4]  = (_Float16)a1.x; a[5]  = (_Float16)a1.y; a[6]  = (_Float16)a1.z; a[7]  = (_Float16)a1.w;
    a[8]  = (_Float16)a2.x; a[9]  = (_Float16)a2.y; a[10] = (_Float16)a2.z; a[11] = (_Float16)a2.w;
    a[12] = (_Float16)a3.x; a[13] = (_Float16)a3.y; a[14] = (_Float16)a3.z; a[15] = (_Float16)a3.w;
    // B 32x16 f16: lane's column, K rows kk+half*16 .. +15 -> contiguous in Wt
    const v16h b0 = *(const v16h*)(bp0 + kk);
    const v16h b1 = *(const v16h*)(bp1 + kk);
    acc0 = __builtin_amdgcn_wmma_f32_16x16x32_f16(false, a, false, b0, (short)0, acc0, false, false);
    acc1 = __builtin_amdgcn_wmma_f32_16x16x32_f16(false, a, false, b1, (short)0, acc1, false, false);
  }
  // D 16x16 f32: VGPR r -> row r (lanes 0-15) / row r+8 (lanes 16-31); no M guard needed
  const int row0 = (tm << 4) + half * 8;
  float* c0 = C + (size_t)row0 * NC + colB0;
  const float bv0 = bias[colB0];
  const float bv1 = bias[colB1];
#pragma unroll
  for (int r = 0; r < 8; ++r) {
    c0[r * NC]      = acc0[r] + bv0;
    c0[r * NC + 16] = acc1[r] + bv1;
  }
}

// ---------- edge pass A: attention logits + segment max (wave32 per edge) ----------

__global__ void edge_alpha_kernel(const int* __restrict__ src, const int* __restrict__ dst,
                                  const float* __restrict__ ea, const float* __restrict__ eamean,
                                  const float* __restrict__ xl, const float* __restrict__ xr,
                                  const float* __restrict__ We, const float* __restrict__ att,
                                  float* __restrict__ alpha, float* __restrict__ amax,
                                  int E, int EE, int heads) {
  int e = blockIdx.x * (blockDim.x >> 5) + (threadIdx.x >> 5);
  if (e >= EE) return;
  const int lane = threadIdx.x & 31;
  int s, d; float eav;
  if (e < E) { s = src[e]; d = dst[e]; eav = ea[e]; }
  else       { s = d = e - E;          eav = eamean[s]; }   // self-loop, fill='mean'
  const int Cn = heads * 32;
  for (int h = 0; h < heads; ++h) {
    const int ch = h * 32 + lane;                 // HID==32 == wave32: lane is channel
    float v = xl[(size_t)s * Cn + ch] + xr[(size_t)d * Cn + ch] + eav * We[ch];
    v = (v > 0.0f) ? v : NEG_SLOPE * v;           // leaky_relu
    float p = v * att[ch];
#pragma unroll
    for (int off = 16; off > 0; off >>= 1) p += __shfl_xor(p, off, 32);
    if (lane == 0) {
      alpha[(size_t)e * heads + h] = p;
      atomicMaxF(&amax[(size_t)d * heads + h], p);
    }
  }
}

// ---------- edge pass B: exp(alpha - max) stored in place, segment sum ----------

__global__ void expsum_kernel(const int* __restrict__ dst, float* __restrict__ alpha,
                              const float* __restrict__ amax, float* __restrict__ denom,
                              int E, int EE, int heads) {
  int idx = blockIdx.x * blockDim.x + threadIdx.x;
  if (idx >= EE * heads) return;
  int e = idx / heads, h = idx - e * heads;
  int d = (e < E) ? dst[e] : (e - E);
  float ex = __expf(alpha[idx] - amax[(size_t)d * heads + h]);
  alpha[idx] = ex;                                  // reuse buffer: ealpha
  atomicAdd(&denom[(size_t)d * heads + h], ex);
}

// ---------- edge pass C: weighted scatter-add of xl[src] ----------

__global__ void scatter_kernel(const int* __restrict__ src, const int* __restrict__ dst,
                               const float* __restrict__ xl, const float* __restrict__ alpha,
                               const float* __restrict__ denom, float* __restrict__ acc,
                               int E, int EE, int heads) {
  int e = blockIdx.x * (blockDim.x >> 5) + (threadIdx.x >> 5);
  if (e >= EE) return;
  const int lane = threadIdx.x & 31;
  int s, d;
  if (e < E) { s = src[e]; d = dst[e]; } else { s = d = e - E; }
  const int Cn = heads * 32;
  for (int h = 0; h < heads; ++h) {
    float coef = alpha[(size_t)e * heads + h] / denom[(size_t)d * heads + h];
    atomicAdd(&acc[(size_t)d * Cn + h * 32 + lane],
              coef * xl[(size_t)s * Cn + h * 32 + lane]);
  }
}

// ---------- bias (+optional ELU), in place ----------

__global__ void bias_act_kernel(float* __restrict__ buf, const float* __restrict__ bias,
                                int n, int Cn, int do_elu) {
  int i = blockIdx.x * blockDim.x + threadIdx.x;
  if (i >= n) return;
  float v = buf[i] + bias[i & (Cn - 1)];            // Cn is 128 or 32 (power of two)
  if (do_elu) v = (v > 0.0f) ? v : (__expf(v) - 1.0f);
  buf[i] = v;
}

// ---------- final head: preds[n, 0..3] = shared[n, :] @ W_out + b_out ----------

__global__ void preds_kernel(const float* __restrict__ sh, const float* __restrict__ Wo,
                             const float* __restrict__ bo, float* __restrict__ out, int n) {
  int i = blockIdx.x * blockDim.x + threadIdx.x;
  if (i >= n) return;
  float s[32];
#pragma unroll
  for (int c = 0; c < 32; ++c) s[c] = sh[(size_t)i * 32 + c];
#pragma unroll
  for (int j = 0; j < 4; ++j) {
    float a = bo[j];
#pragma unroll
    for (int c = 0; c < 32; ++c) a += s[c] * Wo[c * 4 + j];
    out[(size_t)j * n + i] = a;                     // [pred_ad|pred_pd|pred_ftd|pred_als]
  }
}

// ---------- host launch ----------

extern "C" void kernel_launch(void* const* d_in, const int* in_sizes, int n_in,
                              void* d_out, int out_size, void* d_ws, size_t ws_size,
                              hipStream_t stream) {
  (void)n_in; (void)out_size; (void)ws_size;
  const float* x     = (const float*)d_in[0];
  const float* eattr = (const float*)d_in[1];
  const float* Wl1   = (const float*)d_in[2];
  const float* bl1   = (const float*)d_in[3];
  const float* Wr1   = (const float*)d_in[4];
  const float* br1   = (const float*)d_in[5];
  const float* We1   = (const float*)d_in[6];
  const float* att1  = (const float*)d_in[7];
  const float* bias1 = (const float*)d_in[8];
  const float* Wl2   = (const float*)d_in[9];
  const float* bl2   = (const float*)d_in[10];
  const float* Wr2   = (const float*)d_in[11];
  const float* br2   = (const float*)d_in[12];
  const float* We2   = (const float*)d_in[13];
  const float* att2  = (const float*)d_in[14];
  const float* bias2 = (const float*)d_in[15];
  const float* Wout  = (const float*)d_in[16];
  const float* bout  = (const float*)d_in[17];
  const int*   ei    = (const int*)d_in[18];

  const int N  = in_sizes[0] / 128;
  const int E  = in_sizes[1];
  const int EE = E + N;                 // edges + self-loops
  const int* srcI = ei;
  const int* dstI = ei + E;

  float* out        = (float*)d_out;
  float* shared_out = out + 4 * (size_t)N;

  // workspace carve (64-float aligned)
  float* ws = (float*)d_ws;
  size_t off = 0;
  auto carve = [&](size_t nel) { float* p = ws + off; off += (nel + 63) & ~(size_t)63; return p; };
  float* xl1    = carve((size_t)N * 128);
  float* xr1    = carve((size_t)N * 128);
  float* hbuf   = carve((size_t)N * 128);
  float* xl2    = carve((size_t)N * 32);
  float* xr2    = carve((size_t)N * 32);
  float* deg    = carve((size_t)N);
  float* eamean = carve((size_t)N);
  float* alpha1 = carve((size_t)EE * 4);
  float* amax1  = carve((size_t)N * 4);
  float* denom1 = carve((size_t)N * 4);
  float* alpha2 = carve((size_t)EE);
  float* amax2  = carve((size_t)N);
  float* denom2 = carve((size_t)N);
  // f16 repacked weights (carved in float units)
  _Float16* Wt1l = (_Float16*)carve((128 * 128) / 2);
  _Float16* Wt1r = (_Float16*)carve((128 * 128) / 2);
  _Float16* Wt2l = (_Float16*)carve((128 * 32) / 2);
  _Float16* Wt2r = (_Float16*)carve((128 * 32) / 2);

  const int BT = 256;
  auto nb = [&](int n) { return (n + BT - 1) / BT; };
  const int eb = (EE + 7) / 8;          // 8 waves per 256-thread block

  // preprocessing: degree + mean edge_attr (self-loop fill value)
  fill_kernel<<<nb(N), BT, 0, stream>>>(deg, 0.0f, N);
  fill_kernel<<<nb(N), BT, 0, stream>>>(eamean, 0.0f, N);
  deg_easum_kernel<<<nb(E), BT, 0, stream>>>(dstI, eattr, deg, eamean, E);
  eamean_kernel<<<nb(N), BT, 0, stream>>>(eamean, deg, N);

  // weight repacks (tiny)
  repack_w_kernel<<<nb(128 * 128), BT, 0, stream>>>(Wl1, Wt1l, 128, 128);
  repack_w_kernel<<<nb(128 * 128), BT, 0, stream>>>(Wr1, Wt1r, 128, 128);
  repack_w_kernel<<<nb(128 * 32),  BT, 0, stream>>>(Wl2, Wt2l, 128, 32);
  repack_w_kernel<<<nb(128 * 32),  BT, 0, stream>>>(Wr2, Wt2r, 128, 32);

  // ---- layer 1 (4 heads x 32) ----
  {
    int tiles = (N / 16) * (128 / 32);  // 16x32 tile per wave
    gemm_bias_wmma<128, 128><<<(tiles + 7) / 8, 256, 0, stream>>>(x, Wt1l, bl1, xl1, N);
    gemm_bias_wmma<128, 128><<<(tiles + 7) / 8, 256, 0, stream>>>(x, Wt1r, br1, xr1, N);
  }
  fill_kernel<<<nb(4 * N), BT, 0, stream>>>(amax1, -INFINITY, 4 * N);
  fill_kernel<<<nb(4 * N), BT, 0, stream>>>(denom1, 0.0f, 4 * N);
  edge_alpha_kernel<<<eb, 256, 0, stream>>>(srcI, dstI, eattr, eamean, xl1, xr1,
                                            We1, att1, alpha1, amax1, E, EE, 4);
  expsum_kernel<<<nb(EE * 4), BT, 0, stream>>>(dstI, alpha1, amax1, denom1, E, EE, 4);
  fill_kernel<<<nb(N * 128), BT, 0, stream>>>(hbuf, 0.0f, N * 128);
  scatter_kernel<<<eb, 256, 0, stream>>>(srcI, dstI, xl1, alpha1, denom1, hbuf, E, EE, 4);
  bias_act_kernel<<<nb(N * 128), BT, 0, stream>>>(hbuf, bias1, N * 128, 128, 1); // +bias, ELU

  // ---- layer 2 (1 head x 32) ----
  {
    int tiles = (N / 16) * (32 / 32);
    gemm_bias_wmma<128, 32><<<(tiles + 7) / 8, 256, 0, stream>>>(hbuf, Wt2l, bl2, xl2, N);
    gemm_bias_wmma<128, 32><<<(tiles + 7) / 8, 256, 0, stream>>>(hbuf, Wt2r, br2, xr2, N);
  }
  fill_kernel<<<nb(N), BT, 0, stream>>>(amax2, -INFINITY, N);
  fill_kernel<<<nb(N), BT, 0, stream>>>(denom2, 0.0f, N);
  edge_alpha_kernel<<<eb, 256, 0, stream>>>(srcI, dstI, eattr, eamean, xl2, xr2,
                                            We2, att2, alpha2, amax2, E, EE, 1);
  expsum_kernel<<<nb(EE), BT, 0, stream>>>(dstI, alpha2, amax2, denom2, E, EE, 1);
  fill_kernel<<<nb(N * 32), BT, 0, stream>>>(shared_out, 0.0f, N * 32);
  scatter_kernel<<<eb, 256, 0, stream>>>(srcI, dstI, xl2, alpha2, denom2, shared_out, E, EE, 1);
  bias_act_kernel<<<nb(N * 32), BT, 0, stream>>>(shared_out, bias2, N * 32, 32, 0); // +bias only

  // ---- output head ----
  preds_kernel<<<nb(N), BT, 0, stream>>>(shared_out, Wout, bout, out, N);
}
